// EPropRSNU_64768106824235
// MI455X (gfx1250) — matrix operand order repaired and checked by reading, not compiled, not run on hardware
//
#include <hip/hip_runtime.h>

typedef __attribute__((ext_vector_type(16))) _Float16 v16h;
typedef __attribute__((ext_vector_type(8)))  float    v8f;

#define B_ 32
#define I_ 256
#define H_ 512
#define T_ 64

// exp(-1/10), exp(-1/70)
#define ALPHA_ 0.90483741803595957f
#define RHO_   0.98581584235727442f

__device__ __forceinline__ v8f wmma_f16(v16h a, v16h b, v8f c) {
  // D = A(16x32 f16) * B(32x16 f16) + C(16x16 f32)
  return __builtin_amdgcn_wmma_f32_16x16x32_f16(
      /*neg_a=*/false, a, /*neg_b=*/false, b,
      /*c_mod=*/(short)0, c, /*reuse_a=*/false, /*reuse_b=*/false);
}

// ---------------------------------------------------------------------------
// Pack w_ih (I x H) and w_hh (H x H, diagonal zeroed) into WMMA B-fragment
// layout, f16. Fragment = 32 lanes x 16 halves (1 KB), lane L half h holds
// B[k = (h&7) + 8*(L>>4) + 16*(h>>3)][n = L&15] of the 32x16 tile.
// frag 0..255   : w_ih, frag = ntile*8  + kstep   (ntile 0..31, kstep 0..7)
// frag 256..767 : w_hh, frag = ntile*16 + kstep   (ntile 0..31, kstep 0..15)
// ---------------------------------------------------------------------------
__global__ void pack_w_kernel(const float* __restrict__ wih,
                              const float* __restrict__ whh,
                              v16h* __restrict__ out_ih,
                              v16h* __restrict__ out_hh) {
  int tid  = blockIdx.x * blockDim.x + threadIdx.x;   // 768*32 threads
  int frag = tid >> 5;
  int lane = tid & 31;
  int koff = 8 * (lane >> 4);
  int nloc = lane & 15;
  v16h v;
  if (frag < 256) {
    int ntile = frag >> 3, kstep = frag & 7;
    int c = 16 * ntile + nloc;
#pragma unroll
    for (int h = 0; h < 16; ++h) {
      int r = 32 * kstep + koff + (h & 7) + 16 * (h >> 3);  // r < 256 = I_
      v[h] = (_Float16)wih[r * H_ + c];
    }
    out_ih[frag * 32 + lane] = v;
  } else {
    int f2 = frag - 256;
    int ntile = f2 >> 4, kstep = f2 & 15;
    int c = 16 * ntile + nloc;
#pragma unroll
    for (int h = 0; h < 16; ++h) {
      int r = 32 * kstep + koff + (h & 7) + 16 * (h >> 3);  // r < 512 = H_
      float w = whh[r * H_ + c];
      v[h] = (_Float16)((r == c) ? 0.f : w);         // zero diagonal (w_hh_nd)
    }
    out_hh[f2 * 32 + lane] = v;
  }
}

// ---------------------------------------------------------------------------
// Pack x (B x I x T, f32) into per-timestep WMMA A-fragments, f16.
// A-frag (16x32): lane L half h holds A[m = L&15][k = (h&7) + 8*(L>>4) + 16*(h>>3)]
// frag index = ((t*2 + mtile)*8 + kstep), 64*2*8 = 1024 frags.
// ---------------------------------------------------------------------------
__global__ void pack_x_kernel(const float* __restrict__ x,
                              v16h* __restrict__ out_x) {
  int tid  = blockIdx.x * blockDim.x + threadIdx.x;   // 1024*32 threads
  int frag = tid >> 5;
  int lane = tid & 31;
  int t     = frag >> 4;
  int rem   = frag & 15;
  int mtile = rem >> 3;
  int kstep = rem & 7;
  int b  = 16 * mtile + (lane & 15);
  int k8 = 8 * (lane >> 4);
  v16h v;
#pragma unroll
  for (int h = 0; h < 16; ++h) {
    int i = 32 * kstep + (h & 7) + k8 + 16 * (h >> 3);
    v[h] = (_Float16)x[(b * I_ + i) * T_ + t];
  }
  out_x[frag * 32 + lane] = v;
}

// ---------------------------------------------------------------------------
// Persistent recurrence kernel. gridDim.x = 2 (one M=16 batch tile each),
// blockDim.x = 1024 (32 waves). Wave w owns H columns [16w, 16w+16).
// Per step: 8 WMMA (x@w_ih, K=256) + 16 WMMA (spike@w_hh, K=512), then the
// pointwise LIF/adaptive-threshold update; spikes exchanged through a
// double-buffered LDS A-fragment buffer, one workgroup barrier per step.
// Weight fragments are re-loaded from global every step (L0/L2-resident).
// An opaque *scalar* offset (value 0, in SGPR) defeats LICM so the loads are
// never hoisted/spilled, while keeping global addrspace inference intact
// (pure global_load_b128, no flat loads -> no false DScnt coupling with the
// spike LDS traffic).
// ---------------------------------------------------------------------------
__global__ __launch_bounds__(1024) void rsnu_kernel(
    const float* __restrict__ v0, float* __restrict__ out,
    const v16h* __restrict__ wihf, const v16h* __restrict__ whhf,
    const v16h* __restrict__ xf) {
  __shared__ v16h sbuf[2][512];                 // 2 x 16 frags x 32 lanes = 32 KB

  const int tid   = threadIdx.x;
  const int wave  = tid >> 5;
  const int lane  = tid & 31;
  const int mtile = blockIdx.x;
  const int nloc  = lane & 15;
  const int mbase = 8 * (lane >> 4);            // C/D layout: m = j + mbase
  const int hcol  = wave * 16 + nloc;

  // spike_0 = 0 : zero-init buffer 0
  if (tid < 512) {
    v16h z = {};
    sbuf[0][tid] = z;
  }
  __syncthreads();

  // per-lane recurrent state in C/D layout (8 rows per lane)
  float v[8], a[8];
#pragma unroll
  for (int j = 0; j < 8; ++j) {
    int b = mtile * 16 + mbase + j;
    v[j] = v0[b * H_ + hcol];
    a[j] = 0.f;
  }

  const v16h* wih_base = wihf + wave * 8  * 32 + lane;   // 8 B-frags / step
  const v16h* whh_base = whhf + wave * 16 * 32 + lane;   // 16 B-frags / step

  // Constants for scattering this wave's spike tile into A-fragment layout:
  // n_global = 16*wave + nloc -> frag fw = wave>>1, k' = 16*(wave&1)+nloc
  const int kp      = 16 * (wave & 1) + nloc;
  const int fw      = wave >> 1;
  const int half_t  = (kp & 7) + 8 * (kp >> 4);
  const int lane_hi = 16 * ((kp >> 3) & 1);

  for (int t = 0; t < T_; ++t) {
    // Opaque scalar byte-offset (really 0): defeats LICM on the weight loads
    // without destroying global-addrspace inference on the pointers.
    long wofs = 0;
    asm volatile("" : "+s"(wofs));
    const v16h* wih = (const v16h*)((const char*)wih_base + wofs);
    const v16h* whh = (const v16h*)((const char*)whh_base + wofs);

    v8f acc = {};

    // input current: x_t @ w_ih   (K = 256 -> 8 WMMA)
    const v16h* xa = xf + (t * 2 + mtile) * 8 * 32 + lane;
#pragma unroll
    for (int k = 0; k < 8; ++k) {
      v16h af = xa[k * 32];
      v16h bf = wih[k * 32];
      acc = wmma_f16(af, bf, acc);
    }

    // recurrent current: spike @ w_hh_nd   (K = 512 -> 16 WMMA)
    const v16h* sp = &sbuf[t & 1][lane];
#pragma unroll
    for (int k = 0; k < 16; ++k) {
      v16h af = sp[k * 32];
      v16h bf = whh[k * 32];
      acc = wmma_f16(af, bf, acc);
    }

    _Float16* sdst = (_Float16*)&sbuf[(t + 1) & 1][fw * 32];
    float* op = out + (size_t)t * (B_ * H_) + (mtile * 16 + mbase) * H_ + hcol;

#pragma unroll
    for (int j = 0; j < 8; ++j) {
      float vt = ALPHA_ * v[j] + acc[j] - ((v[j] > a[j]) ? a[j] : 0.f);
      float st = (vt > a[j]) ? 1.f : 0.f;     // compare vs OLD threshold state
      a[j] = RHO_ * a[j] + st;
      v[j] = vt;
      // scatter spike (exact in f16) into next step's A-fragment buffer
      int lt = mbase + j + lane_hi;
      sdst[lt * 16 + half_t] = (_Float16)st;
      // emit spikes output (T,B,H) f32 — coalesced across lanes 0..15
      op[j * H_] = st;
    }
    __syncthreads();   // publishes sbuf[(t+1)&1]; also fences reuse of sbuf[t&1]
  }
}

extern "C" void kernel_launch(void* const* d_in, const int* in_sizes, int n_in,
                              void* d_out, int out_size, void* d_ws, size_t ws_size,
                              hipStream_t stream) {
  (void)in_sizes; (void)n_in; (void)out_size; (void)ws_size;
  const float* x   = (const float*)d_in[0];   // (B, I, T)
  const float* v0  = (const float*)d_in[1];   // (B, H)
  const float* wih = (const float*)d_in[2];   // (I, H)
  const float* whh = (const float*)d_in[3];   // (H, H)
  float* out = (float*)d_out;                 // (T, B, H)

  char* ws = (char*)d_ws;
  v16h* wihf = (v16h*)(ws);                   // 256 KB
  v16h* whhf = (v16h*)(ws + (256 << 10));     // 512 KB
  v16h* xf   = (v16h*)(ws + (768 << 10));     // 1 MB   (total 1.75 MB)

  pack_w_kernel<<<96, 256, 0, stream>>>(wih, whh, wihf, whhf);   // 768 frags
  pack_x_kernel<<<128, 256, 0, stream>>>(x, xf);                 // 1024 frags
  rsnu_kernel<<<2, 1024, 0, stream>>>(v0, out, wihf, whhf, xf);
}
